// LocalDecoder_4320737100316
// MI455X (gfx1250) — matrix operand order repaired and checked by reading, not compiled, not run on hardware
//
#include <hip/hip_runtime.h>
#include <stdint.h>

// ---------------------------------------------------------------------------
// Types
// ---------------------------------------------------------------------------
typedef __attribute__((ext_vector_type(4)))  float  f32x4;
typedef __attribute__((ext_vector_type(8)))  float  v8f;
typedef __attribute__((ext_vector_type(4)))  __bf16 v4bf;
typedef __attribute__((ext_vector_type(8)))  __bf16 v8bf;
typedef __attribute__((ext_vector_type(16))) __bf16 v16bf;

#define B_   4
#define S_   1024
#define P_   128
#define GD_  4096
#define D_   768
#define H_   12
#define DH_  64
#define L_   6
#define V_   256
#define FF_  3072

__device__ __forceinline__ __bf16 f2bf(float f) {
  unsigned u = __builtin_bit_cast(unsigned, f);
  u += 0x7FFFu + ((u >> 16) & 1u);            // round-to-nearest-even
  unsigned short h = (unsigned short)(u >> 16);
  return __builtin_bit_cast(__bf16, h);
}

// low 32 bits of a generic pointer to a __shared__ object = LDS byte offset
__device__ __forceinline__ unsigned lds_off(const void* p) {
  return (unsigned)(uintptr_t)p;
}

// gfx1250 async global->LDS copy of 16 bytes per lane (ASYNCcnt-tracked)
__device__ __forceinline__ void async_cp16(unsigned lds_byte, const void* gaddr) {
  asm volatile("global_load_async_to_lds_b128 %0, %1, off"
               :: "v"(lds_byte), "v"((unsigned long long)(uintptr_t)gaddr)
               : "memory");
}
__device__ __forceinline__ void async_wait0() {
  asm volatile("s_wait_asynccnt 0x0" ::: "memory");
}

// assemble a v16bf A/B fragment from two 8-element (16B) LDS chunks
__device__ __forceinline__ v16bf ld16(const __bf16* p0, const __bf16* p1) {
  v8bf a = *(const v8bf*)p0;
  v8bf b = *(const v8bf*)p1;
  v16bf r;
#pragma unroll
  for (int i = 0; i < 8; ++i) { r[i] = a[i]; r[i + 8] = b[i]; }
  return r;
}

// ---------------------------------------------------------------------------
// GEMM (f32 A): C[M,N] = A[M,K] @ W[N,K]^T (+bias)(+relu), C f32 or bf16.
// Block tile 128x128, K-step 32, 8 waves (2x4), wave tile 64x32, bf16 WMMA.
// ---------------------------------------------------------------------------
__global__ __launch_bounds__(256) void gemm_f32a_kernel(
    const float* __restrict__ A, const float* __restrict__ W,
    const float* __restrict__ bias, void* __restrict__ Cv,
    int M, int N, int K, int relu, int c_bf16)
{
  __shared__ __align__(16) __bf16 As[128 * 40];
  __shared__ __align__(16) __bf16 Bs[128 * 40];

  const int tid  = threadIdx.x;
  const int lane = tid & 31;
  const int wave = tid >> 5;
  const int wm   = wave >> 2;           // 0..1  (64 rows)
  const int wn   = wave & 3;            // 0..3  (32 cols)
  const int hi16 = (lane >> 4) & 1;
  const int l15  = lane & 15;

  const int m0 = blockIdx.y * 128;
  const int n0 = blockIdx.x * 128;

  v8f acc[4][2];
#pragma unroll
  for (int mi = 0; mi < 4; ++mi)
#pragma unroll
    for (int ni = 0; ni < 2; ++ni)
#pragma unroll
      for (int j = 0; j < 8; ++j) acc[mi][ni][j] = 0.f;

  const int rld = tid >> 3;             // staging row   0..31
  const int kv4 = tid & 7;              // staging f32x4 0..7

  for (int k0 = 0; k0 < K; k0 += 32) {
    if (k0) __syncthreads();
#pragma unroll
    for (int i = 0; i < 4; ++i) {
      int r = rld + i * 32;
      f32x4 a = *(const f32x4*)(A + (size_t)(m0 + r) * K + k0 + kv4 * 4);
      f32x4 w = *(const f32x4*)(W + (size_t)(n0 + r) * K + k0 + kv4 * 4);
      v4bf pa, pw;
#pragma unroll
      for (int e = 0; e < 4; ++e) { pa[e] = f2bf(a[e]); pw[e] = f2bf(w[e]); }
      *(v4bf*)(As + r * 40 + kv4 * 4) = pa;
      *(v4bf*)(Bs + r * 40 + kv4 * 4) = pw;
    }
    if (k0 + 32 < K) {                  // global_prefetch_b8 of next K tile
      __builtin_prefetch(A + (size_t)(m0 + rld) * K + k0 + 32, 0, 0);
      __builtin_prefetch(W + (size_t)(n0 + rld) * K + k0 + 32, 0, 0);
    }
    __syncthreads();

    v16bf af[4];
#pragma unroll
    for (int mi = 0; mi < 4; ++mi) {
      const __bf16* p = As + (wm * 64 + mi * 16 + l15) * 40 + (hi16 ? 8 : 0);
      af[mi] = ld16(p, p + 16);
    }
    v16bf bfr[2];
#pragma unroll
    for (int ni = 0; ni < 2; ++ni) {
      const __bf16* p = Bs + (wn * 32 + ni * 16 + l15) * 40 + (hi16 ? 8 : 0);
      bfr[ni] = ld16(p, p + 16);
    }
#pragma unroll
    for (int mi = 0; mi < 4; ++mi)
#pragma unroll
      for (int ni = 0; ni < 2; ++ni)
        acc[mi][ni] = __builtin_amdgcn_wmma_f32_16x16x32_bf16(
            false, af[mi], false, bfr[ni], (short)0, acc[mi][ni], false, false);
  }

  if (c_bf16) {
    __bf16* C = (__bf16*)Cv;
#pragma unroll
    for (int ni = 0; ni < 2; ++ni) {
      int n = n0 + wn * 32 + ni * 16 + l15;
      float bv = bias ? bias[n] : 0.f;
#pragma unroll
      for (int mi = 0; mi < 4; ++mi) {
        int mb = m0 + wm * 64 + mi * 16 + hi16 * 8;
#pragma unroll
        for (int j = 0; j < 8; ++j) {
          float v = acc[mi][ni][j] + bv;
          if (relu) v = fmaxf(v, 0.f);
          C[(size_t)(mb + j) * N + n] = f2bf(v);
        }
      }
    }
  } else {
    float* C = (float*)Cv;
#pragma unroll
    for (int ni = 0; ni < 2; ++ni) {
      int n = n0 + wn * 32 + ni * 16 + l15;
      float bv = bias ? bias[n] : 0.f;
#pragma unroll
      for (int mi = 0; mi < 4; ++mi) {
        int mb = m0 + wm * 64 + mi * 16 + hi16 * 8;
#pragma unroll
        for (int j = 0; j < 8; ++j) {
          float v = acc[mi][ni][j] + bv;
          if (relu) v = fmaxf(v, 0.f);
          C[(size_t)(mb + j) * N + n] = v;
        }
      }
    }
  }
}

// ---------------------------------------------------------------------------
// GEMM (bf16 A, async-to-LDS staged): C = A_bf16 @ W_f32^T (+bias)(+relu).
// A tile staging uses GLOBAL_LOAD_ASYNC_TO_LDS_B128 (pure byte copy).
// ---------------------------------------------------------------------------
__global__ __launch_bounds__(256) void gemm_bf16a_kernel(
    const __bf16* __restrict__ A, const float* __restrict__ W,
    const float* __restrict__ bias, void* __restrict__ Cv,
    int M, int N, int K, int relu, int c_bf16)
{
  __shared__ __align__(16) __bf16 As[128 * 40];
  __shared__ __align__(16) __bf16 Bs[128 * 40];

  const int tid  = threadIdx.x;
  const int lane = tid & 31;
  const int wave = tid >> 5;
  const int wm   = wave >> 2;
  const int wn   = wave & 3;
  const int hi16 = (lane >> 4) & 1;
  const int l15  = lane & 15;

  const int m0 = blockIdx.y * 128;
  const int n0 = blockIdx.x * 128;
  const unsigned As_base = lds_off(&As[0]);

  v8f acc[4][2];
#pragma unroll
  for (int mi = 0; mi < 4; ++mi)
#pragma unroll
    for (int ni = 0; ni < 2; ++ni)
#pragma unroll
      for (int j = 0; j < 8; ++j) acc[mi][ni][j] = 0.f;

  const int rld = tid >> 3;
  const int kv4 = tid & 7;

  for (int k0 = 0; k0 < K; k0 += 32) {
    if (k0) __syncthreads();
    // A tile: 128 rows x 64B = 512 16B-chunks, async copy, 2 chunks/thread
#pragma unroll
    for (int i = 0; i < 2; ++i) {
      int idx = tid + i * 256;
      int r = idx >> 2, c = idx & 3;
      async_cp16(As_base + r * 80 + c * 16,
                 A + (size_t)(m0 + r) * K + k0 + c * 8);
    }
    // W tile: f32 -> bf16 convert
#pragma unroll
    for (int i = 0; i < 4; ++i) {
      int r = rld + i * 32;
      f32x4 w = *(const f32x4*)(W + (size_t)(n0 + r) * K + k0 + kv4 * 4);
      v4bf pw;
#pragma unroll
      for (int e = 0; e < 4; ++e) pw[e] = f2bf(w[e]);
      *(v4bf*)(Bs + r * 40 + kv4 * 4) = pw;
    }
    if (k0 + 32 < K)
      __builtin_prefetch(W + (size_t)(n0 + rld) * K + k0 + 32, 0, 0);
    async_wait0();
    __syncthreads();

    v16bf af[4];
#pragma unroll
    for (int mi = 0; mi < 4; ++mi) {
      const __bf16* p = As + (wm * 64 + mi * 16 + l15) * 40 + (hi16 ? 8 : 0);
      af[mi] = ld16(p, p + 16);
    }
    v16bf bfr[2];
#pragma unroll
    for (int ni = 0; ni < 2; ++ni) {
      const __bf16* p = Bs + (wn * 32 + ni * 16 + l15) * 40 + (hi16 ? 8 : 0);
      bfr[ni] = ld16(p, p + 16);
    }
#pragma unroll
    for (int mi = 0; mi < 4; ++mi)
#pragma unroll
      for (int ni = 0; ni < 2; ++ni)
        acc[mi][ni] = __builtin_amdgcn_wmma_f32_16x16x32_bf16(
            false, af[mi], false, bfr[ni], (short)0, acc[mi][ni], false, false);
  }

  if (c_bf16) {
    __bf16* C = (__bf16*)Cv;
#pragma unroll
    for (int ni = 0; ni < 2; ++ni) {
      int n = n0 + wn * 32 + ni * 16 + l15;
      float bv = bias ? bias[n] : 0.f;
#pragma unroll
      for (int mi = 0; mi < 4; ++mi) {
        int mb = m0 + wm * 64 + mi * 16 + hi16 * 8;
#pragma unroll
        for (int j = 0; j < 8; ++j) {
          float v = acc[mi][ni][j] + bv;
          if (relu) v = fmaxf(v, 0.f);
          C[(size_t)(mb + j) * N + n] = f2bf(v);
        }
      }
    }
  } else {
    float* C = (float*)Cv;
#pragma unroll
    for (int ni = 0; ni < 2; ++ni) {
      int n = n0 + wn * 32 + ni * 16 + l15;
      float bv = bias ? bias[n] : 0.f;
#pragma unroll
      for (int mi = 0; mi < 4; ++mi) {
        int mb = m0 + wm * 64 + mi * 16 + hi16 * 8;
#pragma unroll
        for (int j = 0; j < 8; ++j) {
          float v = acc[mi][ni][j] + bv;
          if (relu) v = fmaxf(v, 0.f);
          C[(size_t)(mb + j) * N + n] = v;
        }
      }
    }
  }
}

// ---------------------------------------------------------------------------
// Fused flash attention (bf16 in, bf16 out, WMMA, online softmax).
// Block = 4 waves = 64 queries of one (b,h); key blocks of 32.
// Q/K staged with async-to-LDS byte copies; V transposed via direct bf16 copy.
// ---------------------------------------------------------------------------
__global__ __launch_bounds__(128) void attn_wmma_kernel(
    const __bf16* __restrict__ Q, long long q_bstride, int q_rstride,
    const __bf16* __restrict__ Kp, const __bf16* __restrict__ Vp,
    long long kv_bstride, int kv_rstride,
    __bf16* __restrict__ O, int Lk)
{
  __shared__ __align__(16) __bf16 Qs[4 * 16 * 72];
  __shared__ __align__(16) __bf16 Ks[32 * 72];
  __shared__ __align__(16) __bf16 Vt[64 * 40];
  __shared__ __align__(16) __bf16 Ps[4 * 16 * 40];

  const int tid  = threadIdx.x;
  const int lane = tid & 31;
  const int wave = tid >> 5;
  const int hi16 = (lane >> 4) & 1;
  const int l15  = lane & 15;

  const int bh = blockIdx.y;
  const int b  = bh / H_;
  const int h  = bh % H_;
  const int q0 = blockIdx.x * 64 + wave * 16;

  const __bf16* qb = Q  + (size_t)b * q_bstride  + h * DH_;
  const __bf16* kb = Kp + (size_t)b * kv_bstride + h * DH_;
  const __bf16* vb = Vp + (size_t)b * kv_bstride + h * DH_;

  // stage this wave's 16x64 Q tile: 128 16B chunks, async
  __bf16* qs = Qs + wave * (16 * 72);
  const unsigned qs_base = lds_off(qs);
#pragma unroll
  for (int t = 0; t < 4; ++t) {
    int i   = lane + t * 32;
    int row = i >> 3;
    int c   = i & 7;
    async_cp16(qs_base + row * 144 + c * 16,
               qb + (size_t)(q0 + row) * q_rstride + c * 8);
  }
  async_wait0();

  v16bf qf[2];
#pragma unroll
  for (int ks = 0; ks < 2; ++ks) {
    const __bf16* p = qs + l15 * 72 + ks * 32 + (hi16 ? 8 : 0);
    qf[ks] = ld16(p, p + 16);
  }

  float mrun[8], lrun[8];
  v8f oacc[4];
#pragma unroll
  for (int j = 0; j < 8; ++j) { mrun[j] = -1e30f; lrun[j] = 0.f; }
#pragma unroll
  for (int t = 0; t < 4; ++t)
#pragma unroll
    for (int j = 0; j < 8; ++j) oacc[t][j] = 0.f;

  __bf16* ps = Ps + wave * (16 * 40);
  const unsigned ks_base = lds_off(&Ks[0]);

  for (int kblk = 0; kblk < Lk; kblk += 32) {
    __syncthreads();                    // protect K/V LDS from previous use
    // K block 32x64: 256 16B chunks, async byte copy
#pragma unroll
    for (int t = 0; t < 2; ++t) {
      int i   = tid + t * 128;
      int row = i >> 3;
      int c   = i & 7;
      async_cp16(ks_base + row * 144 + c * 16,
                 kb + (size_t)(kblk + row) * kv_rstride + c * 8);
    }
    // V block transposed: Vt[dh][key], direct bf16 copy
#pragma unroll
    for (int t = 0; t < 16; ++t) {
      int i   = tid + t * 128;
      int d   = i & 63;
      int key = i >> 6;
      Vt[d * 40 + key] = vb[(size_t)(kblk + key) * kv_rstride + d];
    }
    async_wait0();
    __syncthreads();

    // scores S = Q @ K^T : 16 queries x 32 keys
    v8f sc[2];
#pragma unroll
    for (int nt = 0; nt < 2; ++nt) {
#pragma unroll
      for (int j = 0; j < 8; ++j) sc[nt][j] = 0.f;
#pragma unroll
      for (int ks = 0; ks < 2; ++ks) {
        const __bf16* p = Ks + (nt * 16 + l15) * 72 + ks * 32 + (hi16 ? 8 : 0);
        v16bf kf = ld16(p, p + 16);
        sc[nt] = __builtin_amdgcn_wmma_f32_16x16x32_bf16(
            false, qf[ks], false, kf, (short)0, sc[nt], false, false);
      }
    }

    // online softmax (rows in VGPR index j; 16-lane half reductions)
#pragma unroll
    for (int j = 0; j < 8; ++j) {
      float s0 = sc[0][j] * 0.125f;     // 1/sqrt(64)
      float s1 = sc[1][j] * 0.125f;
      float mx = fmaxf(s0, s1);
#pragma unroll
      for (int m = 1; m < 16; m <<= 1) mx = fmaxf(mx, __shfl_xor(mx, m, 32));
      float mnew = fmaxf(mrun[j], mx);
      float p0 = __expf(s0 - mnew);
      float p1 = __expf(s1 - mnew);
      float rs = p0 + p1;
#pragma unroll
      for (int m = 1; m < 16; m <<= 1) rs += __shfl_xor(rs, m, 32);
      float alpha = __expf(mrun[j] - mnew);
      lrun[j] = lrun[j] * alpha + rs;
      mrun[j] = mnew;
#pragma unroll
      for (int t = 0; t < 4; ++t) oacc[t][j] *= alpha;
      int mrow = j + hi16 * 8;
      ps[mrow * 40 + l15]      = f2bf(p0);
      ps[mrow * 40 + 16 + l15] = f2bf(p1);
    }

    // O += P @ V
    {
      const __bf16* pp = ps + l15 * 40 + (hi16 ? 8 : 0);
      v16bf pf = ld16(pp, pp + 16);
#pragma unroll
      for (int t = 0; t < 4; ++t) {
        const __bf16* vp = Vt + (t * 16 + l15) * 40 + (hi16 ? 8 : 0);
        v16bf vf = ld16(vp, vp + 16);
        oacc[t] = __builtin_amdgcn_wmma_f32_16x16x32_bf16(
            false, pf, false, vf, (short)0, oacc[t], false, false);
      }
    }
  }

  // write O (bf16) = oacc / l
#pragma unroll
  for (int t = 0; t < 4; ++t) {
    int d = h * DH_ + t * 16 + l15;
#pragma unroll
    for (int j = 0; j < 8; ++j) {
      int qrow = q0 + j + hi16 * 8;
      O[((size_t)b * S_ + qrow) * D_ + d] = f2bf(oacc[t][j] / lrun[j]);
    }
  }
}

// ---------------------------------------------------------------------------
// x = LayerNorm(x + y) * g + b   (one block per row of 768, f32)
// ---------------------------------------------------------------------------
__global__ __launch_bounds__(256) void add_ln_kernel(
    float* __restrict__ x, const float* __restrict__ y,
    const float* __restrict__ g, const float* __restrict__ bb)
{
  const int row  = blockIdx.x;
  const int tid  = threadIdx.x;
  const int lane = tid & 31;
  const int wave = tid >> 5;
  const size_t base = (size_t)row * D_;

  float s[3];
  float sum = 0.f, sq = 0.f;
#pragma unroll
  for (int c = 0; c < 3; ++c) {
    int col = tid + c * 256;
    float v = x[base + col] + y[base + col];
    s[c] = v; sum += v; sq += v * v;
  }
#pragma unroll
  for (int m = 1; m < 32; m <<= 1) {
    sum += __shfl_xor(sum, m, 32);
    sq  += __shfl_xor(sq,  m, 32);
  }
  __shared__ float r1[8], r2[8];
  if (lane == 0) { r1[wave] = sum; r2[wave] = sq; }
  __syncthreads();
  sum = 0.f; sq = 0.f;
#pragma unroll
  for (int i = 0; i < 8; ++i) { sum += r1[i]; sq += r2[i]; }
  float mean = sum * (1.f / D_);
  float var  = sq * (1.f / D_) - mean * mean;
  float inv  = rsqrtf(var + 1e-5f);
#pragma unroll
  for (int c = 0; c < 3; ++c) {
    int col = tid + c * 256;
    x[base + col] = (s[c] - mean) * inv * g[col] + bb[col];
  }
}

// ---------------------------------------------------------------------------
// x[b,s,:] = src[b*P + patch_ids[b,s], :]   (f32)
// ---------------------------------------------------------------------------
__global__ __launch_bounds__(256) void gather_kernel(
    const float* __restrict__ src, const int* __restrict__ pid,
    float* __restrict__ dst)
{
  const int row = blockIdx.x;          // b*S + s
  const int b   = row >> 10;           // S = 1024
  const int p   = pid[row];
  const float* sp = src + (size_t)(b * P_ + p) * D_;
  float* dp = dst + (size_t)row * D_;
  for (int c = threadIdx.x; c < D_; c += 256) dp[c] = sp[c];
}

// ---------------------------------------------------------------------------
// Host launcher
// ---------------------------------------------------------------------------
static inline void gemmF(const float* A, const float* W, const float* bias,
                         void* C, int M, int N, int K, int relu, int cbf,
                         hipStream_t s) {
  gemm_f32a_kernel<<<dim3(N / 128, M / 128), 256, 0, s>>>(A, W, bias, C, M, N, K, relu, cbf);
}
static inline void gemmB(const __bf16* A, const float* W, const float* bias,
                         void* C, int M, int N, int K, int relu, int cbf,
                         hipStream_t s) {
  gemm_bf16a_kernel<<<dim3(N / 128, M / 128), 256, 0, s>>>(A, W, bias, C, M, N, K, relu, cbf);
}

extern "C" void kernel_launch(void* const* d_in, const int* in_sizes, int n_in,
                              void* d_out, int out_size, void* d_ws, size_t ws_size,
                              hipStream_t stream) {
  (void)in_sizes; (void)n_in; (void)out_size; (void)ws_size;

  const float* patch  = (const float*)d_in[0];
  /* d_in[1] = encoder_hidden_states : unused by the reference forward */
  const int*   pid    = (const int*)  d_in[2];
  const float* inW    = (const float*)d_in[3];
  const float* inB    = (const float*)d_in[4];
  const float* Wv     = (const float*)d_in[5];
  const float* Wo     = (const float*)d_in[6];
  const float* saqkvW = (const float*)d_in[7];
  const float* saqkvB = (const float*)d_in[8];
  const float* saoW   = (const float*)d_in[9];
  const float* saoB   = (const float*)d_in[10];
  const float* caqkvW = (const float*)d_in[11];
  const float* caqkvB = (const float*)d_in[12];
  const float* caoW   = (const float*)d_in[13];
  const float* caoB   = (const float*)d_in[14];
  const float* f1W    = (const float*)d_in[15];
  const float* f1B    = (const float*)d_in[16];
  const float* f2W    = (const float*)d_in[17];
  const float* f2B    = (const float*)d_in[18];
  const float* l1g    = (const float*)d_in[19];
  const float* l1b    = (const float*)d_in[20];
  const float* l2g    = (const float*)d_in[21];
  const float* l2b    = (const float*)d_in[22];
  const float* l3g    = (const float*)d_in[23];
  const float* l3b    = (const float*)d_in[24];
  const float* outW   = (const float*)d_in[25];
  const float* outB   = (const float*)d_in[26];

  const int MP = B_ * P_;   // 512 patch rows
  const int MS = B_ * S_;   // 4096 byte rows

  // f32 region, then bf16 region
  float* wsf  = (float*)d_ws;
  float* proj = wsf;                         // [512, 768]   f32
  float* x    = proj + (size_t)MP * D_;      // [4096, 768]  f32
  float* y    = x    + (size_t)MS * D_;      // [4096, 768]  f32
  float* tmp  = y    + (size_t)MS * D_;      // [512, 768]   f32
  __bf16* bigb  = (__bf16*)(tmp + (size_t)MP * D_);  // [4096, 3072] bf16
  __bf16* attnb = bigb  + (size_t)MS * FF_;          // [4096, 768]  bf16
  __bf16* kvb   = attnb + (size_t)MS * D_;           // [512, 1536]  bf16

  // ---- patch-level preamble: proj, (proj@Wv^T)@Wo^T, gather ----
  gemmF(patch, inW, inB, proj, MP, D_, GD_, 0, 0, stream);
  gemmF(proj, Wv, nullptr, y,   MP, D_, D_, 0, 0, stream);   // t = proj@Wv^T
  gemmF(y, Wo, nullptr, tmp,    MP, D_, D_, 0, 0, stream);   // proj2 = t@Wo^T
  gather_kernel<<<MS, 256, 0, stream>>>(tmp, pid, x);

  const dim3 attn_grid(S_ / 64, B_ * H_);

  for (int i = 0; i < L_; ++i) {
    // ---- self attention ----
    gemmF(x, saqkvW + (size_t)i * 3 * D_ * D_, saqkvB + (size_t)i * 3 * D_,
          bigb, MS, 3 * D_, D_, 0, 1, stream);               // qkv bf16 [4096,2304]
    attn_wmma_kernel<<<attn_grid, 128, 0, stream>>>(
        bigb,               (long long)S_ * 3 * D_, 3 * D_,
        bigb + D_, bigb + 2 * D_, (long long)S_ * 3 * D_, 3 * D_,
        attnb, S_);
    gemmB(attnb, saoW + (size_t)i * D_ * D_, saoB + (size_t)i * D_,
          y, MS, D_, D_, 0, 0, stream);
    add_ln_kernel<<<MS, 256, 0, stream>>>(x, y, l1g + i * D_, l1b + i * D_);

    // ---- cross attention (memory = proj, 512 rows) ----
    gemmF(x, caqkvW + (size_t)i * 3 * D_ * D_, caqkvB + (size_t)i * 3 * D_,
          bigb, MS, D_, D_, 0, 1, stream);                   // q bf16 [4096,768]
    gemmF(proj, caqkvW + (size_t)i * 3 * D_ * D_ + (size_t)D_ * D_,
          caqkvB + (size_t)i * 3 * D_ + D_,
          kvb, MP, 2 * D_, D_, 0, 1, stream);                // k|v bf16 [512,1536]
    attn_wmma_kernel<<<attn_grid, 128, 0, stream>>>(
        bigb,        (long long)S_ * D_, D_,
        kvb, kvb + D_, (long long)P_ * 2 * D_, 2 * D_,
        attnb, P_);
    gemmB(attnb, caoW + (size_t)i * D_ * D_, caoB + (size_t)i * D_,
          y, MS, D_, D_, 0, 0, stream);
    add_ln_kernel<<<MS, 256, 0, stream>>>(x, y, l2g + i * D_, l2b + i * D_);

    // ---- FFN ----
    gemmF(x, f1W + (size_t)i * FF_ * D_, f1B + (size_t)i * FF_,
          bigb, MS, FF_, D_, 1, 1, stream);                  // h1 bf16 (relu)
    gemmB(bigb, f2W + (size_t)i * D_ * FF_, f2B + (size_t)i * D_,
          y, MS, D_, FF_, 0, 0, stream);
    add_ln_kernel<<<MS, 256, 0, stream>>>(x, y, l3g + i * D_, l3b + i * D_);
  }

  // ---- output head (f32 logits) ----
  gemmF(x, outW, outB, d_out, MS, V_, D_, 0, 0, stream);
}
